// MultiHeadSelfAttention2D_76905684402750
// MI455X (gfx1250) — matrix operand order repaired and checked by reading, not compiled, not run on hardware
//
#include <hip/hip_runtime.h>
#include <hip/hip_bf16.h>

// MI455X / gfx1250, wave32. Flash-attention MHSA using v_wmma_f32_16x16x32_f16.
// Shapes fixed by the reference: B=2, C=256, H=W=64 -> N=4096, HEADS=8, HD=32.
//
// All WMMA operand layouts are chosen so each lane's 16 fragment halves form
// two contiguous 8-half runs (global_load_b128 / ds_load_b128 pairs, no scalar gathers).
// flash_attn stages K/V chunks block-cooperatively in double-buffered LDS,
// via GLOBAL_LOAD_ASYNC_TO_LDS (ASYNCcnt) when the toolchain exposes it.

typedef __attribute__((ext_vector_type(16))) _Float16 v16h;
typedef __attribute__((ext_vector_type(8)))  float    v8f;
typedef int v4i __attribute__((vector_size(16)));
typedef __attribute__((address_space(1))) v4i gv4i;
typedef __attribute__((address_space(3))) v4i lv4i;

#define BATCH 2
#define CCH   256
#define NTOK  4096
#define HEADS 8
#define HD    32
#define OCH   (3 * CCH)   // 768

#if defined(__has_builtin)
#if __has_builtin(__builtin_amdgcn_global_load_async_to_lds_b128) && \
    __has_builtin(__builtin_amdgcn_s_wait_asynccnt)
#define USE_ASYNC_COPY 1
#endif
#endif

// ---- WMMA per-lane fragment index maps (CDNA5 ISA 7.12.2, wave32) ----
// A matrix (16-bit, 16xK32): lane holds row M = lane&15; half i -> K below.
__device__ __forceinline__ int kmapA(int lane, int i) {
  int h8 = (lane & 16) ? 8 : 0;
  return (i < 8) ? (h8 + i) : (16 + h8 + (i - 8));   // two contiguous 8-runs
}
// B matrix (16-bit, 32x16): lane holds col N = lane&15; half i -> K below.
__device__ __forceinline__ int kmapB(int lane, int i) {
  int h8 = (lane & 16) ? 8 : 0;
  return (i < 8) ? (h8 + i) : (16 + h8 + (i - 8));   // two contiguous 8-runs
}
// C/D f32 16x16: reg r -> row = r + ((lane&16)?8:0); col = lane&15.

__device__ __forceinline__ v8f wmma_f16(v16h a, v16h b, v8f c) {
  return __builtin_amdgcn_wmma_f32_16x16x32_f16(false, a, false, b, (short)0, c,
                                                false, false);
}

// ---------------- Kernel 0a: plain f32 -> f16 convert (weights) -------------
__global__ void cvt_f32_to_f16(const float* __restrict__ src,
                               _Float16* __restrict__ dst, int n) {
  int i = blockIdx.x * blockDim.x + threadIdx.x;
  int stride = gridDim.x * blockDim.x;
  for (; i < n; i += stride) dst[i] = (_Float16)src[i];
}

// ---------------- Kernel 0b: x [b][c][n] f32 -> xT [b][n][c] f16 ------------
// grid (NTOK/256, BATCH*CCH); coalesced reads, one-time scattered writes.
__global__ void cvt_transpose_x(const float* __restrict__ src,
                                _Float16* __restrict__ dst) {
  int n = blockIdx.x * blockDim.x + threadIdx.x;
  int bc = blockIdx.y;
  int b = bc >> 8, c = bc & 255;
  dst[((size_t)b * NTOK + n) * CCH + c] = (_Float16)src[(size_t)bc * NTOK + n];
}

// ---------------- Kernel 1: QKV projection ----------------
// qkv[b] (768 x N) = w_qkv (768 x 256) @ x[b] (256 x N) + b_qkv
// One wave computes a 16x64 strip: A-frag reused 4x, 4 independent WMMA chains.
// Outputs scattered to attention-friendly layouts:
//   q,k -> token-major [b*H + h][n][d] ; v -> channel-major [b*H + h][d][n]
__global__ void qkv_gemm(const _Float16* __restrict__ wq,
                         const _Float16* __restrict__ xt,
                         const float* __restrict__ bias,
                         _Float16* __restrict__ qT,
                         _Float16* __restrict__ kT,
                         _Float16* __restrict__ vC) {
  const int lane = threadIdx.x;
  const int mo = blockIdx.y * 8 + threadIdx.y;  // 0..47 (row tiles of 768)
  const int n0 = blockIdx.x * 64;               // col strip start
  const int b  = blockIdx.z;
  const _Float16* xb = xt + (size_t)b * NTOK * CCH;
  v8f acc[4] = {{}, {}, {}, {}};
  for (int k0 = 0; k0 < CCH; k0 += 32) {
    v16h a;
#pragma unroll
    for (int i = 0; i < 16; ++i)
      a[i] = wq[(size_t)(mo * 16 + (lane & 15)) * CCH + k0 + kmapA(lane, i)];
#pragma unroll
    for (int t = 0; t < 4; ++t) {
      v16h bm;
#pragma unroll
      for (int i = 0; i < 16; ++i)
        bm[i] = xb[(size_t)(n0 + t * 16 + (lane & 15)) * CCH + k0 + kmapB(lane, i)];
      acc[t] = wmma_f16(a, bm, acc[t]);
    }
  }
  // Epilogue: which/head/dbase are WAVE-UNIFORM (a 16-row tile never straddles
  // a q/k/v or head boundary) -> scalar-ish branches, minimal divergence.
  const int which = mo >> 4;          // 0=q, 1=k, 2=v
  const int h     = (mo & 15) >> 1;   // head
  const int dbase = (mo & 1) * 16;    // d offset of this tile within the head
  const int half8 = (lane & 16) ? 8 : 0;
  const size_t bh = (size_t)b * HEADS + h;
  if (which < 2) {
    _Float16* dst = (which == 0 ? qT : kT) + bh * NTOK * HD;  // [n][d]
#pragma unroll
    for (int r = 0; r < 8; ++r) {
      int d = dbase + r + half8;
      float bs = bias[mo * 16 + r + half8];
#pragma unroll
      for (int t = 0; t < 4; ++t) {
        int n = n0 + t * 16 + (lane & 15);
        dst[(size_t)n * HD + d] = (_Float16)(acc[t][r] + bs);
      }
    }
  } else {
    _Float16* dst = vC + bh * HD * NTOK;                       // [d][n]
#pragma unroll
    for (int r = 0; r < 8; ++r) {
      int d = dbase + r + half8;
      float bs = bias[mo * 16 + r + half8];
#pragma unroll
      for (int t = 0; t < 4; ++t) {
        int n = n0 + t * 16 + (lane & 15);
        dst[(size_t)d * NTOK + n] = (_Float16)(acc[t][r] + bs);
      }
    }
  }
}

// ---------------- Kernel 2: flash attention per (b, head) ----------------
// blockDim = (32, 8): 8 waves x 16 query rows = 128 query rows per block.
// K/V key-chunks (32 tokens) are staged block-cooperatively into LDS with
// double buffering; fragments then come from ds_load (two 8-half runs).
__global__ void flash_attn(const _Float16* __restrict__ qT,
                           const _Float16* __restrict__ kT,
                           const _Float16* __restrict__ vC,
                           _Float16* __restrict__ aT) {
  const int lane = threadIdx.x;
  const int wv = threadIdx.y;
  const int tid = wv * 32 + lane;             // 0..255
  const int n0 = (blockIdx.x * 8 + wv) * 16;  // query tile start
  const int hd = blockIdx.y;                  // head
  const int b  = blockIdx.z;

  __shared__ __align__(16) _Float16 kbuf[2][32 * 32];  // [token][d], 2KB each
  __shared__ __align__(16) _Float16 vbuf[2][32 * 32];  // [d][m],     2KB each
  __shared__ __align__(16) _Float16 pbuf[8][16 * 32];  // wave-private P tile

  const size_t bh = (size_t)b * HEADS + hd;
  const _Float16* qb = qT + bh * NTOK * HD;   // [n][d]
  const _Float16* kb = kT + bh * NTOK * HD;   // [m][d]  (chunk = 2KB contig)
  const _Float16* vb = vC + bh * HD * NTOK;   // [d][m]  (chunk = 32 x 64B rows)

  // Q fragment (A layout): row = query n, K = d (two contiguous 8-half runs).
  v16h qa;
#pragma unroll
  for (int i = 0; i < 16; ++i)
    qa[i] = qb[(size_t)(n0 + (lane & 15)) * HD + kmapA(lane, i)];

  // Cooperative chunk loader: 256 threads x 16B = 4KB (K 2KB + V 2KB).
  auto coop_load = [&](int buf, int m0) {
#ifdef USE_ASYNC_COPY
    if (tid < 128) {
      __builtin_amdgcn_global_load_async_to_lds_b128(
          (gv4i*)(kb + (size_t)m0 * HD + tid * 8),
          (lv4i*)(&kbuf[buf][tid * 8]), 0, 0);
    } else {
      int j = tid - 128;
      int d = j >> 2, q = j & 3;
      __builtin_amdgcn_global_load_async_to_lds_b128(
          (gv4i*)(vb + (size_t)d * NTOK + m0 + q * 8),
          (lv4i*)(&vbuf[buf][j * 8]), 0, 0);
    }
#else
    if (tid < 128) {
      const uint4* src = (const uint4*)(kb + (size_t)m0 * HD);
      ((uint4*)kbuf[buf])[tid] = src[tid];
    } else {
      int j = tid - 128;
      int d = j >> 2, q = j & 3;
      const uint4* src = (const uint4*)(vb + (size_t)d * NTOK + m0);
      ((uint4*)vbuf[buf])[j] = src[q];
    }
#endif
  };

  coop_load(0, 0);
#ifdef USE_ASYNC_COPY
  __builtin_amdgcn_s_wait_asynccnt(0);
#endif
  __syncthreads();

  float mi[8], li[8];
  v8f o0 = {}, o1 = {};  // O^T tile: cols d=0..15 and d=16..31
#pragma unroll
  for (int r = 0; r < 8; ++r) { mi[r] = -3.0e38f; li[r] = 0.f; }
  const float scale = 0.17677669529663687f;  // 1/sqrt(32)

  for (int m0 = 0; m0 < NTOK; m0 += 32) {
    const int cur = (m0 >> 5) & 1;
    if (m0 + 32 < NTOK) {
      coop_load(cur ^ 1, m0 + 32);  // overlap next chunk copy with compute
      __builtin_prefetch(kb + (size_t)(m0 + 64) * HD + tid * 8, 0, 2);
    }
    // K fragments from LDS (B layout): col = key token, K = d.
    v16h bk0, bk1;
#pragma unroll
    for (int i = 0; i < 16; ++i) {
      bk0[i] = kbuf[cur][(lane & 15) * 32 + kmapB(lane, i)];
      bk1[i] = kbuf[cur][(16 + (lane & 15)) * 32 + kmapB(lane, i)];
    }
    v8f z = {};
    v8f s0 = wmma_f16(qa, bk0, z);
    v8f s1 = wmma_f16(qa, bk1, z);

    // Online softmax. A row's 16 columns live across one 16-lane half,
    // so row reductions are width-16 xor shuffles.
#pragma unroll
    for (int r = 0; r < 8; ++r) {
      float a0 = s0[r] * scale, a1 = s1[r] * scale;
      float t = fmaxf(a0, a1);
      t = fmaxf(t, __shfl_xor(t, 1, 16));
      t = fmaxf(t, __shfl_xor(t, 2, 16));
      t = fmaxf(t, __shfl_xor(t, 4, 16));
      t = fmaxf(t, __shfl_xor(t, 8, 16));
      float mn = fmaxf(mi[r], t);
      float al = __expf(mi[r] - mn);
      float p0 = __expf(a0 - mn);
      float p1 = __expf(a1 - mn);
      float rs = p0 + p1;
      rs += __shfl_xor(rs, 1, 16);
      rs += __shfl_xor(rs, 2, 16);
      rs += __shfl_xor(rs, 4, 16);
      rs += __shfl_xor(rs, 8, 16);
      li[r] = li[r] * al + rs;
      mi[r] = mn;
      o0[r] *= al;
      o1[r] *= al;
      // Stash P tile (C/D layout -> LDS row-major 16x32) for A-layout reload.
      int row = r + ((lane & 16) ? 8 : 0);
      pbuf[wv][row * 32 + (lane & 15)] = (_Float16)p0;
      pbuf[wv][row * 32 + 16 + (lane & 15)] = (_Float16)p1;
    }
    // Reload P as A fragment; V fragments from LDS (same-wave DS in order).
    v16h pa, bv0, bv1;
#pragma unroll
    for (int i = 0; i < 16; ++i) {
      pa[i]  = pbuf[wv][(lane & 15) * 32 + kmapA(lane, i)];
      bv0[i] = vbuf[cur][(lane & 15) * 32 + kmapB(lane, i)];         // d 0..15
      bv1[i] = vbuf[cur][(16 + (lane & 15)) * 32 + kmapB(lane, i)];  // d 16..31
    }
    o0 = wmma_f16(pa, bv0, o0);
    o1 = wmma_f16(pa, bv1, o1);
    // All waves done with `cur`; next-buffer copies complete before reuse.
#ifdef USE_ASYNC_COPY
    __builtin_amdgcn_s_wait_asynccnt(0);
#endif
    __syncthreads();
  }

  // Normalize; store token-major f16 [b][n][c] for the out-proj GEMM B side.
  _Float16* ab = aT + (size_t)b * NTOK * CCH;
  const int half8 = (lane & 16) ? 8 : 0;
#pragma unroll
  for (int r = 0; r < 8; ++r) {
    float inv = 1.0f / li[r];
    int n = n0 + r + half8;
    ab[(size_t)n * CCH + hd * HD + (lane & 15)]      = (_Float16)(o0[r] * inv);
    ab[(size_t)n * CCH + hd * HD + 16 + (lane & 15)] = (_Float16)(o1[r] * inv);
  }
}

// ---------------- Kernel 3: output projection + residual ----------------
// out[b] = gamma * (w_out (256x256) @ attn[b] (256 x N) + b_out) + x[b]
__global__ void out_proj(const _Float16* __restrict__ wo,
                         const _Float16* __restrict__ aT,
                         const float* __restrict__ bo,
                         const float* __restrict__ x,
                         const float* __restrict__ gamma,
                         float* __restrict__ out) {
  const int lane = threadIdx.x;
  const int mo = blockIdx.y * 8 + threadIdx.y;  // 0..15
  const int n0 = blockIdx.x * 64;
  const int b  = blockIdx.z;
  const _Float16* ab = aT + (size_t)b * NTOK * CCH;  // [n][c]
  v8f acc[4] = {{}, {}, {}, {}};
  for (int k0 = 0; k0 < CCH; k0 += 32) {
    v16h a;
#pragma unroll
    for (int i = 0; i < 16; ++i)
      a[i] = wo[(size_t)(mo * 16 + (lane & 15)) * CCH + k0 + kmapA(lane, i)];
#pragma unroll
    for (int t = 0; t < 4; ++t) {
      v16h bm;
#pragma unroll
      for (int i = 0; i < 16; ++i)
        bm[i] = ab[(size_t)(n0 + t * 16 + (lane & 15)) * CCH + k0 + kmapB(lane, i)];
      acc[t] = wmma_f16(a, bm, acc[t]);
    }
  }
  const float g = gamma[0];
  const int half8 = (lane & 16) ? 8 : 0;
  const float* xb = x + (size_t)b * CCH * NTOK;
  float* ob = out + (size_t)b * CCH * NTOK;
#pragma unroll
  for (int r = 0; r < 8; ++r) {
    int row = mo * 16 + r + half8;
    float bs = bo[row];
#pragma unroll
    for (int t = 0; t < 4; ++t) {
      int col = n0 + t * 16 + (lane & 15);
      ob[(size_t)row * NTOK + col] =
          g * (acc[t][r] + bs) + xb[(size_t)row * NTOK + col];
    }
  }
}

extern "C" void kernel_launch(void* const* d_in, const int* in_sizes, int n_in,
                              void* d_out, int out_size, void* d_ws, size_t ws_size,
                              hipStream_t stream) {
  const float* x      = (const float*)d_in[0];  // B*C*N
  const float* w_qkv  = (const float*)d_in[1];  // 768*256
  const float* b_qkv  = (const float*)d_in[2];  // 768
  const float* w_out  = (const float*)d_in[3];  // 256*256
  const float* b_out  = (const float*)d_in[4];  // 256
  const float* gamma  = (const float*)d_in[5];  // 1
  float* out = (float*)d_out;

  // Workspace carve-up (f16 elements): ~21.5 MB total.
  const size_t PLANE = (size_t)BATCH * CCH * NTOK;            // 2,097,152
  _Float16* xt    = (_Float16*)d_ws;                          // B*N*C
  _Float16* wqkvh = xt    + PLANE;                            // 768*256
  _Float16* wouth = wqkvh + (size_t)OCH * CCH;                // 256*256
  _Float16* qT    = wouth + (size_t)CCH * CCH;                // B*H*N*HD
  _Float16* kT    = qT + PLANE;
  _Float16* vC    = kT + PLANE;
  _Float16* aT    = vC + PLANE;                               // B*N*C

  {
    dim3 grid(NTOK / 256, BATCH * CCH);
    cvt_transpose_x<<<grid, 256, 0, stream>>>(x, xt);
  }
  cvt_f32_to_f16<<<192, 256, 0, stream>>>(w_qkv, wqkvh, OCH * CCH);
  cvt_f32_to_f16<<<64, 256, 0, stream>>>(w_out, wouth, CCH * CCH);

  {
    dim3 grid(NTOK / 64, OCH / (16 * 8), BATCH);   // (64, 6, 2)
    dim3 block(32, 8);
    qkv_gemm<<<grid, block, 0, stream>>>(wqkvh, xt, b_qkv, qT, kT, vC);
  }
  {
    dim3 grid(NTOK / (16 * 8), HEADS, BATCH);      // (32, 8, 2)
    dim3 block(32, 8);
    flash_attn<<<grid, block, 0, stream>>>(qT, kT, vC, aT);
  }
  {
    dim3 grid(NTOK / 64, CCH / (16 * 8), BATCH);   // (64, 2, 2)
    dim3 block(32, 8);
    out_proj<<<grid, block, 0, stream>>>(wouth, aT, b_out, x, gamma, out);
  }
}
